// GCN_3607772529222
// MI455X (gfx1250) — compile-verified, matrix operand order checked by reading, not run on hardware
//
#include <hip/hip_runtime.h>
#include <hip/hip_bf16.h>
#include <math.h>

#define N_NODES 50000
#define N_EDGES 800000
#define NFEAT   512
#define NHID    96
#define NCLASS  40
#define MTILES  (N_NODES / 16)          // 3125, exact
#define KSTEPS  (NFEAT / 32)            // 16
#define NTILES  (NHID / 16)             // 6
#define W1P_ELEMS (KSTEPS * NTILES * 32 * 16)   // 49152 halfs

typedef __attribute__((ext_vector_type(16))) _Float16 v16h;
typedef __attribute__((ext_vector_type(8)))  float    v8f;

// ---------------------------------------------------------------------------
// Repack W1 (f32, row-major [K=512][N=96]) into f16 B-fragment layout so each
// lane's 16 halfs for one 32x16 K-tile are contiguous (one 32-byte load).
// ISA 16-bit B layout (lane = column n, K packed): lane L: n = L%16, khalf=L/16,
// elem i<8 -> k = khalf*8+i ; i>=8 -> k = 16 + khalf*8 + (i-8), within a k-step.
// ---------------------------------------------------------------------------
__global__ __launch_bounds__(256) void k_prep_w1(const float* __restrict__ W1,
                                                 _Float16* __restrict__ w1p) {
    int p = blockIdx.x * 256 + threadIdx.x;
    if (p >= W1P_ELEMS) return;
    int i     = p & 15;
    int lane  = (p >> 4) & 31;
    int nt    = (p >> 9) % NTILES;
    int kstep = (p >> 9) / NTILES;
    int n     = nt * 16 + (lane & 15);
    int khalf = lane >> 4;
    int k     = kstep * 32 + ((i < 8) ? (khalf * 8 + i) : (16 + khalf * 8 + (i - 8)));
    w1p[p] = (_Float16)W1[k * NHID + n];
}

// ---------------------------------------------------------------------------
// GEMM1: support[50000][96] = x @ W1 + b1 via v_wmma_f32_16x16x32_f16.
// One wave per 16-row M-tile, full N (6 accumulators), K-loop of 16 steps.
// ---------------------------------------------------------------------------
__global__ __launch_bounds__(256) void k_gemm1(const float* __restrict__ x,
                                               const _Float16* __restrict__ w1p,
                                               const float* __restrict__ b1,
                                               float* __restrict__ support) {
    int tid  = blockIdx.x * 256 + threadIdx.x;
    int wave = tid >> 5;
    int lane = tid & 31;
    if (wave >= MTILES) return;              // whole-wave exit: EXEC all-1s for WMMA

    int khalf = lane >> 4;
    int nl    = lane & 15;
    int m     = wave * 16 + nl;              // A-fragment row owned by this lane

    v8f acc[NTILES] = {};
    const v16h* bp = (const v16h*)w1p;

    for (int ks = 0; ks < KSTEPS; ++ks) {
        int k0 = ks * 32 + khalf * 8;
        const float* xr = x + m * NFEAT + k0;
        v8f c0 = *(const v8f*)xr;            // K = k0 .. k0+7      (32B aligned)
        v8f c1 = *(const v8f*)(xr + 16);     // K = k0+16 .. k0+23  (32B aligned)
        v16h a;
        #pragma unroll
        for (int i = 0; i < 8; ++i) {
            a[i]     = (_Float16)c0[i];
            a[i + 8] = (_Float16)c1[i];
        }
        #pragma unroll
        for (int nt = 0; nt < NTILES; ++nt) {
            v16h b = bp[(ks * NTILES + nt) * 32 + lane];
            acc[nt] = __builtin_amdgcn_wmma_f32_16x16x32_f16(
                false, a, false, b, (short)0, acc[nt], false, false);
        }
    }

    // C layout: lane n = lane&15; rows M = khalf*8 + r in VGPR r.
    int mbase = wave * 16 + khalf * 8;
    #pragma unroll
    for (int nt = 0; nt < NTILES; ++nt) {
        int n = nt * 16 + nl;
        float bias = b1[n];
        #pragma unroll
        for (int r = 0; r < 8; ++r)
            support[(mbase + r) * NHID + n] = acc[nt][r] + bias;
    }
}

// ---------------------------------------------------------------------------
// Zero agg each call (harness does not re-poison/clear between replays).
// ---------------------------------------------------------------------------
__global__ __launch_bounds__(256) void k_zero(float* __restrict__ p, int n) {
    int i = blockIdx.x * 256 + threadIdx.x;
    if (i < n) p[i] = 0.0f;
}

// ---------------------------------------------------------------------------
// Edge scatter: agg[dst] += support[src] * w. One thread per (edge, feature).
// agg (19 MB) is L2-resident -> relaxed agent-scope f32 atomics resolve in L2.
// ---------------------------------------------------------------------------
__global__ __launch_bounds__(256) void k_scatter(const int* __restrict__ ei,
                                                 const float* __restrict__ ew,
                                                 const float* __restrict__ support,
                                                 float* __restrict__ agg) {
    int idx = blockIdx.x * 256 + threadIdx.x;
    if (idx >= N_EDGES * NHID) return;
    int e = idx / NHID;
    int f = idx - e * NHID;
    int src = ei[e];
    int dst = ei[N_EDGES + e];
    float v = support[src * NHID + f] * ew[e];
    __hip_atomic_fetch_add(&agg[dst * NHID + f], v,
                           __ATOMIC_RELAXED, __HIP_MEMORY_SCOPE_AGENT);
}

// ---------------------------------------------------------------------------
// ReLU + logits = h @ W2 + b2 + log_softmax, fused. One thread per row;
// W2 (15 KB) + b2 staged in LDS, uniform LDS reads broadcast across the wave.
// ---------------------------------------------------------------------------
__global__ __launch_bounds__(128) void k_out(const float* __restrict__ agg,
                                             const float* __restrict__ W2,
                                             const float* __restrict__ b2,
                                             float* __restrict__ out) {
    __shared__ float W2s[NHID * NCLASS];
    __shared__ float b2s[NCLASS];
    for (int i = threadIdx.x; i < NHID * NCLASS; i += 128) W2s[i] = W2[i];
    if (threadIdx.x < NCLASS) b2s[threadIdx.x] = b2[threadIdx.x];
    __syncthreads();

    int r = blockIdx.x * 128 + threadIdx.x;
    if (r >= N_NODES) return;

    float lg[NCLASS];
    #pragma unroll
    for (int c = 0; c < NCLASS; ++c) lg[c] = b2s[c];

    const float* hrow = agg + r * NHID;
    for (int k = 0; k < NHID; ++k) {
        float hk = hrow[k];
        hk = hk > 0.0f ? hk : 0.0f;          // ReLU
        #pragma unroll
        for (int c = 0; c < NCLASS; ++c)
            lg[c] = fmaf(hk, W2s[k * NCLASS + c], lg[c]);
    }

    float mx = lg[0];
    #pragma unroll
    for (int c = 1; c < NCLASS; ++c) mx = fmaxf(mx, lg[c]);
    float s = 0.0f;
    #pragma unroll
    for (int c = 0; c < NCLASS; ++c) s += __expf(lg[c] - mx);
    float lse = __logf(s) + mx;
    #pragma unroll
    for (int c = 0; c < NCLASS; ++c) out[r * NCLASS + c] = lg[c] - lse;
}

// ---------------------------------------------------------------------------
extern "C" void kernel_launch(void* const* d_in, const int* in_sizes, int n_in,
                              void* d_out, int out_size, void* d_ws, size_t ws_size,
                              hipStream_t stream) {
    const float* x  = (const float*)d_in[0];
    const int*   ei = (const int*)d_in[1];     // JAX default x64-off -> int32
    const float* ew = (const float*)d_in[2];
    const float* W1 = (const float*)d_in[3];
    const float* b1 = (const float*)d_in[4];
    const float* W2 = (const float*)d_in[5];
    const float* b2 = (const float*)d_in[6];
    float* out = (float*)d_out;

    float*    support = (float*)d_ws;                       // 19.2 MB
    float*    agg     = support + (size_t)N_NODES * NHID;   // 19.2 MB
    _Float16* w1p     = (_Float16*)(agg + (size_t)N_NODES * NHID);  // 96 KB

    k_prep_w1<<<(W1P_ELEMS + 255) / 256, 256, 0, stream>>>(W1, w1p);
    k_zero<<<(N_NODES * NHID + 255) / 256, 256, 0, stream>>>(agg, N_NODES * NHID);
    k_gemm1<<<(MTILES * 32 + 255) / 256, 256, 0, stream>>>(x, w1p, b1, support);
    k_scatter<<<(N_EDGES * NHID + 255) / 256, 256, 0, stream>>>(ei, ew, support, agg);
    k_out<<<(N_NODES + 127) / 128, 128, 0, stream>>>(agg, W2, b2, out);
}
